// TGCN_16363825397959
// MI455X (gfx1250) — compile-verified, matrix operand order checked by reading, not run on hardware
//
#include <hip/hip_runtime.h>
#include <hip/hip_bf16.h>

typedef __attribute__((ext_vector_type(16))) _Float16 v16h;
typedef __attribute__((ext_vector_type(8)))  _Float16 v8h;
typedef __attribute__((ext_vector_type(8)))  float    v8f;

#define Bn   4
#define Nn   10000
#define Tn   16
#define Fn   32
#define Hn   64
#define TOUT 4
#define BT   (Bn*Tn)        // 64 (b,t) slices
#define NT   (Nn/16)        // 625 n-tiles
#define NROW (BT*Nn)        // 640000 GCN gemm rows
#define QROW (Bn*Nn)        // 40000 GRU rows

// Concatenate two contiguous 8-half runs into one WMMA operand.
// CDNA5 16-bit tile layout: lane's halves are K = khalf+{0..7} and khalf+{16..23}
__device__ __forceinline__ v16h cat8(v8h lo, v8h hi) {
    return __builtin_shufflevector(lo, hi, 0, 1, 2, 3, 4, 5, 6, 7,
                                           8, 9, 10, 11, 12, 13, 14, 15);
}
__device__ __forceinline__ v16h ld_op(const _Float16* p) {
    v8h lo = *reinterpret_cast<const v8h*>(p);
    v8h hi = *reinterpret_cast<const v8h*>(p + 16);
    return cat8(lo, hi);
}

// ---------------- CSR build ----------------
__global__ void k_zero_i32(int* p, int n) {
    int i = blockIdx.x * blockDim.x + threadIdx.x;
    if (i < n) p[i] = 0;
}
__global__ void k_zero_f32(float* p, int n) {
    int i = blockIdx.x * blockDim.x + threadIdx.x;
    if (i < n) p[i] = 0.f;
}
__global__ void k_degree(const int* eidx, int E, int* deg) {
    int i = blockIdx.x * blockDim.x + threadIdx.x;
    if (i < E) atomicAdd(&deg[eidx[E + i]], 1);
}
__global__ void k_scan(const int* deg, int* row_off, int* cursor) {
    __shared__ int buf[1024];
    __shared__ int base;
    int tid = threadIdx.x;
    if (tid == 0) base = 0;
    __syncthreads();
    for (int c0 = 0; c0 < Nn; c0 += 1024) {
        int v = (c0 + tid < Nn) ? deg[c0 + tid] : 0;
        buf[tid] = v;
        __syncthreads();
        for (int off = 1; off < 1024; off <<= 1) {
            int t = (tid >= off) ? buf[tid - off] : 0;
            __syncthreads();
            buf[tid] += t;
            __syncthreads();
        }
        if (c0 + tid < Nn) {
            int excl = base + buf[tid] - v;
            row_off[c0 + tid] = excl;
            cursor[c0 + tid]  = excl;
        }
        __syncthreads();
        if (tid == 0) base += buf[1023];
        __syncthreads();
    }
    if (tid == 0) row_off[Nn] = base;
}
__global__ void k_fill(const int* eidx, int E, int* cursor, int* nbr) {
    int i = blockIdx.x * blockDim.x + threadIdx.x;
    if (i < E) {
        int dst = eidx[E + i];
        int pos = atomicAdd(&cursor[dst], 1);
        nbr[pos] = eidx[i];
    }
}

// ------------- weight prep: fp32 -> f16, TRANSPOSED ([col][K]) -------------
__global__ void k_prep_wg(const float* Wself, const float* Wnbr, _Float16* WgT) {
    int i = blockIdx.x * blockDim.x + threadIdx.x;
    if (i < 64 * 64) {
        int c = i >> 6, k = i & 63;     // WgT[c*64 + k] = Wcat[k][c]
        float v = (k < 32) ? Wself[k * 64 + c] : Wnbr[(k - 32) * 64 + c];
        WgT[i] = (_Float16)v;
    }
}
// Fused GRU weight, transposed: 256 cols x 128 K, K = [xt(64) | h(64)],
// cols: [0,64)=r-sum, [64,128)=z-sum, [128,192)=i_n only, [192,256)=h_n only
__global__ void k_prep_wr(const float* Wi, const float* Wh, _Float16* WrT) {
    int i = blockIdx.x * blockDim.x + threadIdx.x;
    if (i < 256 * 128) {
        int c = i >> 7, k = i & 127, g = c >> 6, cc = c & 63;
        float v = 0.f;
        if (g == 0)      v = (k < 64) ? Wi[k * 192 + cc]       : Wh[(k - 64) * 192 + cc];
        else if (g == 1) v = (k < 64) ? Wi[k * 192 + 64 + cc]  : Wh[(k - 64) * 192 + 64 + cc];
        else if (g == 2) v = (k < 64) ? Wi[k * 192 + 128 + cc] : 0.f;
        else             v = (k < 64) ? 0.f                    : Wh[(k - 64) * 192 + 128 + cc];
        WrT[i] = (_Float16)v;
    }
}

// ---------------- mean-neighbor aggregation (CSR gather, L2-resident X) ----
// one node per block; 512 threads = 64 (b,t) rows x 8 float4 feature groups
__global__ void __launch_bounds__(512) k_agg(const float* X, const int* deg,
                                             const int* row_off, const int* nbr,
                                             float* agg) {
    int n   = blockIdx.x;
    int tid = threadIdx.x;
    int row = tid >> 3;          // bt in [0,64)
    int f4  = tid & 7;
    int b = row >> 4, t = row & 15;
    int degn = deg[n];
    int r0   = row_off[n];
    __shared__ int s_idx[256];
    float4 acc = {0.f, 0.f, 0.f, 0.f};
    for (int base = 0; base < degn; base += 256) {
        int cnt = degn - base; if (cnt > 256) cnt = 256;
        if (tid < cnt) s_idx[tid] = nbr[r0 + base + tid];
        __syncthreads();
        for (int i = 0; i < cnt; i++) {
            int src = s_idx[i];
            const float4 xv = *reinterpret_cast<const float4*>(
                X + ((size_t)((b * Nn + src) * Tn + t)) * Fn + f4 * 4);
            if (i + 1 < cnt)
                __builtin_prefetch(X + ((size_t)((b * Nn + s_idx[i + 1]) * Tn + t)) * Fn + f4 * 4);
            acc.x += xv.x; acc.y += xv.y; acc.z += xv.z; acc.w += xv.w;
        }
        __syncthreads();
    }
    float inv = 1.f / (float)((degn > 0) ? degn : 1);
    acc.x *= inv; acc.y *= inv; acc.z *= inv; acc.w *= inv;
    *reinterpret_cast<float4*>(agg + ((size_t)(row * Nn + n)) * Fn + f4 * 4) = acc;
}

// ---------------- GCN: relu([x | mean] @ [Wself;Wnbr] + b), WMMA f16->f32 ---
// 128 threads = 4 waves; each wave: one 16-row M tile x all 64 cols, K=64
__global__ void __launch_bounds__(128) k_gcn(const float* X, const float* agg,
                                             const _Float16* WgT, const float* bgcn,
                                             float* gcn_out) {
    __shared__ _Float16 sWT[64 * 64];        // [col][K]
    __shared__ _Float16 sA[4][16 * 64];      // [wave][M][K]
    int tid = threadIdx.x, lane = tid & 31, wave = tid >> 5;
    for (int i = tid; i < (64 * 64) / 8; i += 128)
        ((uint4*)sWT)[i] = ((const uint4*)WgT)[i];

    int tile = blockIdx.x * 4 + wave;        // [0, 40000)
    int bt = tile / NT, nt = tile % NT;
    int b = bt >> 4, t = bt & 15, n0 = nt * 16;

    for (int idx = lane; idx < 256; idx += 32) {   // 16 rows x 16 float4 groups
        int row = idx >> 4, g = idx & 15;
        float4 v;
        if (g < 8)
            v = *reinterpret_cast<const float4*>(
                X + ((size_t)((b * Nn + n0 + row) * Tn + t)) * Fn + g * 4);
        else
            v = *reinterpret_cast<const float4*>(
                agg + ((size_t)(bt * Nn + n0 + row)) * Fn + (g - 8) * 4);
        union { _Float16 h[4]; uint2 u; } pk;
        pk.h[0] = (_Float16)v.x; pk.h[1] = (_Float16)v.y;
        pk.h[2] = (_Float16)v.z; pk.h[3] = (_Float16)v.w;
        *reinterpret_cast<uint2*>(&sA[wave][row * 64 + g * 4]) = pk.u;
    }
    __syncthreads();

    int khalf = (lane >> 4) * 8;
    int col   = lane & 15;
    v8f c[4] = {};
#pragma unroll
    for (int kb = 0; kb < 64; kb += 32) {
        v16h a = ld_op(&sA[wave][col * 64 + kb + khalf]);      // 2x ds_load_b128
#pragma unroll
        for (int s = 0; s < 4; s++) {
            v16h bm = ld_op(&sWT[(s * 16 + col) * 64 + kb + khalf]);
            c[s] = __builtin_amdgcn_wmma_f32_16x16x32_f16(false, a, false, bm,
                                                          (short)0, c[s], false, false);
        }
    }
#pragma unroll
    for (int s = 0; s < 4; s++) {
        float bb = bgcn[s * 16 + col];
#pragma unroll
        for (int i = 0; i < 8; i++) {
            int m = i + khalf;
            float v = c[s][i] + bb;
            v = v > 0.f ? v : 0.f;
            gcn_out[((size_t)((b * Nn + n0 + m) * Tn + t)) * Hn + s * 16 + col] = v;
        }
    }
}

// ---------------- GRU step: [xt|h] @ Wfused -> gates -> hnew ---------------
// 128 threads = 4 waves over the SAME 16 rows; wave w owns cols [64w,64w+64)
__global__ void __launch_bounds__(128) k_gru(const float* gcn_out, float* hws,
                                             const _Float16* WrT, const float* bi,
                                             const float* bh, int t) {
    __shared__ _Float16 sA[16 * 128];        // [M][K]
    __shared__ float    sS[16 * 256];        // gate pre-activations
    int tid = threadIdx.x, lane = tid & 31, wave = tid >> 5;
    int q0 = blockIdx.x * 16;

    for (int i = tid; i < 512; i += 128) {   // 16 rows x 32 float4 groups
        int row = i >> 5, c4 = i & 31;
        float4 v;
        if (c4 < 16)
            v = *reinterpret_cast<const float4*>(
                gcn_out + ((size_t)((q0 + row) * Tn + t)) * Hn + c4 * 4);
        else
            v = *reinterpret_cast<const float4*>(
                hws + (size_t)(q0 + row) * Hn + (c4 - 16) * 4);
        union { _Float16 h[4]; uint2 u; } pk;
        pk.h[0] = (_Float16)v.x; pk.h[1] = (_Float16)v.y;
        pk.h[2] = (_Float16)v.z; pk.h[3] = (_Float16)v.w;
        *reinterpret_cast<uint2*>(&sA[row * 128 + c4 * 4]) = pk.u;
    }
    __syncthreads();

    int khalf = (lane >> 4) * 8;
    int col   = lane & 15;
    v8f c[4] = {};
#pragma unroll
    for (int kb = 0; kb < 128; kb += 32) {
        v16h a = ld_op(&sA[col * 128 + kb + khalf]);           // 2x ds_load_b128
#pragma unroll
        for (int s = 0; s < 4; s++) {
            int cb = wave * 64 + s * 16 + col;
            v16h bm = ld_op(WrT + (size_t)cb * 128 + kb + khalf);  // 2x global_load_b128
            c[s] = __builtin_amdgcn_wmma_f32_16x16x32_f16(false, a, false, bm,
                                                          (short)0, c[s], false, false);
        }
    }
#pragma unroll
    for (int s = 0; s < 4; s++)
#pragma unroll
        for (int i = 0; i < 8; i++)
            sS[(i + khalf) * 256 + wave * 64 + s * 16 + col] = c[s][i];
    __syncthreads();

    for (int e = tid; e < 1024; e += 128) {  // 16 rows x 64 h
        int row = e >> 6, h = e & 63;
        float r  = sS[row * 256 + h]       + bi[h]       + bh[h];
        r = 1.f / (1.f + __expf(-r));
        float z  = sS[row * 256 + 64 + h]  + bi[64 + h]  + bh[64 + h];
        z = 1.f / (1.f + __expf(-z));
        float nn = sS[row * 256 + 128 + h] + bi[128 + h] +
                   r * (sS[row * 256 + 192 + h] + bh[128 + h]);
        nn = tanhf(nn);
        float hp = hws[(size_t)(q0 + row) * Hn + h];
        hws[(size_t)(q0 + row) * Hn + h] = (1.f - z) * nn + z * hp;
    }
}

// ---------------- output projection (tiny) ---------------------------------
__global__ void k_out(const float* hws, const float* Wo, const float* bo, float* rnn) {
    int i = blockIdx.x * blockDim.x + threadIdx.x;
    if (i < QROW * TOUT) {
        int q = i >> 2, o = i & 3;
        float s = bo[o];
        const float* hp = hws + (size_t)q * Hn;
#pragma unroll 8
        for (int h = 0; h < Hn; h++) s += hp[h] * Wo[h * TOUT + o];
        rnn[i] = s;
    }
}

extern "C" void kernel_launch(void* const* d_in, const int* in_sizes, int n_in,
                              void* d_out, int out_size, void* d_ws, size_t ws_size,
                              hipStream_t stream) {
    (void)n_in; (void)out_size; (void)ws_size;
    const float* X     = (const float*)d_in[0];
    const int*   eidx  = (const int*)d_in[1];
    const float* Wself = (const float*)d_in[2];
    const float* Wnbr  = (const float*)d_in[3];
    const float* bgcn  = (const float*)d_in[4];
    const float* Wi    = (const float*)d_in[5];
    const float* Wh    = (const float*)d_in[6];
    const float* bi    = (const float*)d_in[7];
    const float* bh    = (const float*)d_in[8];
    const float* Wo    = (const float*)d_in[9];
    const float* bo    = (const float*)d_in[10];
    int E = in_sizes[1] / 2;

    float* rnn     = (float*)d_out;
    float* gcn_out = rnn + (size_t)Bn * Nn * TOUT;

    // workspace layout (~93 MB total)
    char*  w   = (char*)d_ws;
    size_t off = 0;
    auto take = [&](size_t bytes) -> char* {
        char* p = w + off;
        off = (off + bytes + 255) & ~(size_t)255;
        return p;
    };
    int*      deg     = (int*)take((size_t)Nn * 4);
    int*      row_off = (int*)take((size_t)(Nn + 1) * 4);
    int*      cursor  = (int*)take((size_t)Nn * 4);
    int*      nbr     = (int*)take((size_t)E * 4);
    _Float16* WgT     = (_Float16*)take(64 * 64 * 2);
    _Float16* WrT     = (_Float16*)take(256 * 128 * 2);
    float*    hws     = (float*)take((size_t)QROW * Hn * 4);
    float*    agg     = (float*)take((size_t)NROW * Fn * 4);

    k_zero_i32<<<(Nn + 255) / 256, 256, 0, stream>>>(deg, Nn);
    k_degree<<<(E + 255) / 256, 256, 0, stream>>>(eidx, E, deg);
    k_scan<<<1, 1024, 0, stream>>>(deg, row_off, cursor);
    k_fill<<<(E + 255) / 256, 256, 0, stream>>>(eidx, E, cursor, nbr);
    k_prep_wg<<<(64 * 64 + 255) / 256, 256, 0, stream>>>(Wself, Wnbr, WgT);
    k_prep_wr<<<(256 * 128 + 255) / 256, 256, 0, stream>>>(Wi, Wh, WrT);
    k_zero_f32<<<(QROW * Hn + 255) / 256, 256, 0, stream>>>(hws, QROW * Hn);

    k_agg<<<Nn, 512, 0, stream>>>(X, deg, row_off, nbr, agg);
    k_gcn<<<Nn, 128, 0, stream>>>(X, agg, WgT, bgcn, gcn_out);  // 10000 blocks x 4 wave-tiles

    for (int t = 0; t < Tn; t++)
        k_gru<<<QROW / 16, 128, 0, stream>>>(gcn_out, hws, WrT, bi, bh, t);

    k_out<<<(QROW * TOUT + 255) / 256, 256, 0, stream>>>(hws, Wo, bo, rnn);
}
